// BiLSTMCRFClassifier_24438363914802
// MI455X (gfx1250) — compile-verified, matrix operand order checked by reading, not run on hardware
//
#include <hip/hip_runtime.h>

// ---------------------------------------------------------------------------
// BiLSTM-CRF for MI455X (gfx1250, wave32).
// GEMMs use v_wmma_f32_16x16x32_bf16 (fp32 accum). Input projections are
// hoisted into one M=8192 WMMA GEMM per layer/dir; the recurrent scan runs one
// small kernel per timestep (16x3072x768 per dir). A-tiles staged via TDM.
// Each wave drives TWO output tiles per k-step from ONE shared A-fragment.
// ---------------------------------------------------------------------------

typedef __attribute__((ext_vector_type(16))) __bf16 v16bf;
typedef __attribute__((ext_vector_type(8)))  float  v8f;
typedef __attribute__((ext_vector_type(4)))  unsigned int u32x4;
typedef __attribute__((ext_vector_type(8)))  int i32x8;
typedef __attribute__((ext_vector_type(4)))  int i32x4;

#ifndef __has_builtin
#define __has_builtin(x) 0
#endif
#if __has_builtin(__builtin_amdgcn_tensor_load_to_lds)
#define HAVE_TDM 1
#else
#define HAVE_TDM 0
#endif

#define Bn 16
#define Tn 512
#define Dn 768
#define Hn 768
#define Kc 9
#define G4 3072   // 4*H

__device__ __forceinline__ unsigned short f2bf(float f) {
  unsigned u = __float_as_uint(f);
  unsigned r = u + 0x7FFFu + ((u >> 16) & 1u);
  return (unsigned short)(r >> 16);
}
__device__ __forceinline__ float sigmoidf_(float x) { return 1.f / (1.f + __expf(-x)); }

union FragU { v16bf v; unsigned u[8]; };

// 16-bit operand fragment load (ISA 7.12.2 layout). Works for both A (row =
// M-row) and B (row = N-row of W[N][K], since B[k][n] = W[n][k] and the
// per-VGPR K pairs are consecutive in memory).
__device__ __forceinline__ void load_frag(FragU& f, const unsigned short* base,
                                          int ld, int row, int k0, int lane) {
  const unsigned short* p = base + (size_t)row * ld + k0 + ((lane >> 4) & 1) * 8;
#pragma unroll
  for (int v = 0; v < 4; ++v) {
    f.u[v]     = *(const unsigned*)(p + 2 * v);
    f.u[v + 4] = *(const unsigned*)(p + 16 + 2 * v);
  }
}

// ---------------------------------------------------------------------------
__global__ void k_lengths(const float* __restrict__ mask, int* __restrict__ len) {
  int b = threadIdx.x;
  if (b < Bn) {
    int s = 0;
    for (int t = 0; t < Tn; ++t) s += (mask[b * Tn + t] > 0.5f);
    len[b] = s;
  }
}

__global__ void k_f2bf(const float* __restrict__ s, unsigned short* __restrict__ d, int n) {
  int i = blockIdx.x * blockDim.x + threadIdx.x;
  if (i < n) d[i] = f2bf(s[i]);
}

__global__ void k_bias(const float* __restrict__ a, const float* __restrict__ b,
                       float* __restrict__ o) {
  int i = blockIdx.x * blockDim.x + threadIdx.x;
  if (i < G4) o[i] = a[i] + b[i];
}

__global__ void k_reverse(const unsigned short* __restrict__ src,
                          unsigned short* __restrict__ dst,
                          const int* __restrict__ len, int C) {
  int idx = blockIdx.x * blockDim.x + threadIdx.x;
  int total = Bn * Tn * C;
  if (idx >= total) return;
  int c = idx % C; int r = idx / C; int t = r % Tn; int b = r / Tn;
  int lb = len[b]; int tr = (t < lb) ? (lb - 1 - t) : t;
  dst[(b * Tn + t) * C + c] = src[(b * Tn + tr) * C + c];
}

__global__ void k_zero_state(unsigned short* __restrict__ hbuf, float* __restrict__ cst) {
  int i = blockIdx.x * blockDim.x + threadIdx.x;
  if (i < 2 * 2 * Bn * Hn) hbuf[i] = 0;
  if (i < 2 * Bn * Hn) cst[i] = 0.f;
}

// ---------------------------------------------------------------------------
// Input-projection GEMM: G[row][4H] = X[row][K] @ W[4H][K]^T + bias, row=b*T+t.
// Grid (512, 12), block 256 (8 waves). Each wave owns two 16x16 tiles (n0 and
// n0+128) sharing one A-fragment per k-step. A tile (16 contiguous rows)
// staged to LDS via the Tensor Data Mover.
__global__ void k_input_gemm(const unsigned short* __restrict__ X,
                             const unsigned short* __restrict__ W,
                             const float* __restrict__ bias,
                             float* __restrict__ G, int Kdim) {
  extern __shared__ unsigned short sA[];
  const int m0 = blockIdx.x * 16;
  const int lane = threadIdx.x & 31;
  const int wave = threadIdx.x >> 5;
  const int n0 = blockIdx.y * 256 + wave * 16;  // second tile at n0+128

#if HAVE_TDM
  if (threadIdx.x < 32) {
    unsigned long long ga = (unsigned long long)(size_t)(X + (size_t)m0 * Kdim);
    unsigned ldsOff = (unsigned)(unsigned long long)(size_t)sA;
    u32x4 g0;
    g0.x = 1u;                                   // count=1, user descriptor
    g0.y = ldsOff;                               // lds_addr
    g0.z = (unsigned)(ga & 0xFFFFFFFFull);       // global_addr lo
    g0.w = (unsigned)((ga >> 32) & 0x01FFFFFFull) | (2u << 30);  // hi + type=2
    unsigned td0 = (unsigned)Kdim, td1 = (unsigned)(Bn * Tn);
    i32x8 g1;
    g1[0] = (int)(1u << 16);                                   // data_size=2B
    g1[1] = (int)((td0 & 0xFFFFu) << 16);                      // tensor_dim0 lo
    g1[2] = (int)((td0 >> 16) | ((td1 & 0xFFFFu) << 16));      // dim0 hi | dim1 lo
    g1[3] = (int)((td1 >> 16) | ((unsigned)Kdim << 16));       // dim1 hi | tile_dim0
    g1[4] = 16;                                                // tile_dim1=16
    g1[5] = (int)td0;                                          // dim0_stride lo
    g1[6] = 0; g1[7] = 0;
    i32x4 z4 = (i32x4)0;
#if defined(__clang_major__) && (__clang_major__ >= 23)
    i32x8 z8 = (i32x8)0;
    __builtin_amdgcn_tensor_load_to_lds(g0, g1, z4, z4, z8, 0);
#else
    __builtin_amdgcn_tensor_load_to_lds(g0, g1, z4, z4, 0);
#endif
#if __has_builtin(__builtin_amdgcn_s_wait_tensorcnt)
    __builtin_amdgcn_s_wait_tensorcnt(0);
#endif
  }
  __syncthreads();
#else
  {
    const unsigned* src = (const unsigned*)(X + (size_t)m0 * Kdim);
    unsigned* dst = (unsigned*)sA;
    int words = 16 * Kdim / 2;
    for (int i = threadIdx.x; i < words; i += blockDim.x) dst[i] = src[i];
  }
  __syncthreads();
#endif

  v8f acc0 = {0.f, 0.f, 0.f, 0.f, 0.f, 0.f, 0.f, 0.f};
  v8f acc1 = {0.f, 0.f, 0.f, 0.f, 0.f, 0.f, 0.f, 0.f};
  const int arow = lane & 15;
  const int brow0 = n0 + (lane & 15);
  const int brow1 = brow0 + 128;
  for (int k0 = 0; k0 < Kdim; k0 += 32) {
    __builtin_prefetch(W + (size_t)brow0 * Kdim + k0 + 256, 0, 3);
    __builtin_prefetch(W + (size_t)brow1 * Kdim + k0 + 256, 0, 3);
    FragU a, b0, b1;
    load_frag(a, sA, Kdim, arow, k0, lane);
    load_frag(b0, W, Kdim, brow0, k0, lane);
    load_frag(b1, W, Kdim, brow1, k0, lane);
    acc0 = __builtin_amdgcn_wmma_f32_16x16x32_bf16(false, a.v, false, b0.v,
                                                   (short)0, acc0, false, false);
    acc1 = __builtin_amdgcn_wmma_f32_16x16x32_bf16(false, a.v, false, b1.v,
                                                   (short)0, acc1, false, false);
  }
  const int n = n0 + (lane & 15);
  const int mb = m0 + ((lane >> 4) & 1) * 8;
  const float bs0 = bias[n];
  const float bs1 = bias[n + 128];
#pragma unroll
  for (int v = 0; v < 8; ++v) {
    G[(size_t)(mb + v) * G4 + n] = acc0[v] + bs0;
    G[(size_t)(mb + v) * G4 + n + 128] = acc1[v] + bs1;
  }
}

// ---------------------------------------------------------------------------
// One LSTM step (both directions). Grid (12, 2), block 256. Each block owns 64
// h-columns: computes the 4 gate slices (i,f,g,o) for them over full K=768 via
// WMMA (two tiles per wave, single shared A-fragment per k-step), then the
// pointwise LSTM cell. h double-buffered (read t&1, write ~t&1) so cross-block
// h reads never race with writes.
__global__ void k_lstm_step(const unsigned short* __restrict__ WhhF,
                            const unsigned short* __restrict__ WhhB,
                            const float* __restrict__ Gf,
                            const float* __restrict__ Gb,
                            unsigned short* __restrict__ hbuf,   // [2][2][16*768]
                            float* __restrict__ cstate,          // [2][16*768]
                            float* __restrict__ houtF,
                            float* __restrict__ houtB,
                            int t) {
  __shared__ unsigned short sh[Bn * Hn];   // h_prev (bf16) 24KB
  __shared__ float sg[4][Bn][64];          // gate slices 16KB
  const int dir = blockIdx.y;
  const unsigned short* Whh = dir ? WhhB : WhhF;
  const float* G = dir ? Gb : Gf;
  float* hout = dir ? houtB : houtF;
  const unsigned short* hread = hbuf + (size_t)((t & 1) * 2 + dir) * (Bn * Hn);
  unsigned short* hwrite = hbuf + (size_t)(((t + 1) & 1) * 2 + dir) * (Bn * Hn);
  float* cst = cstate + (size_t)dir * (Bn * Hn);

  {
    const unsigned* src = (const unsigned*)hread;
    unsigned* dst = (unsigned*)sh;
    for (int i = threadIdx.x; i < Bn * Hn / 2; i += blockDim.x) dst[i] = src[i];
  }
  __syncthreads();

  const int lane = threadIdx.x & 31;
  const int wave = threadIdx.x >> 5;
  const int c0 = blockIdx.x * 64;
  // tiles 2*wave and 2*wave+1 are always in the same gate g (pairs aligned);
  // they share the A matrix (h_prev), so one A-frag feeds two WMMAs.
  const int tile0 = wave * 2;
  const int g = tile0 >> 2, jj = tile0 & 3;          // second tile: jj+1
  const int n0t = g * Hn + c0 + jj * 16;
  const int brow0 = n0t + (lane & 15);
  const int brow1 = brow0 + 16;
  v8f acc0 = {0.f, 0.f, 0.f, 0.f, 0.f, 0.f, 0.f, 0.f};
  v8f acc1 = {0.f, 0.f, 0.f, 0.f, 0.f, 0.f, 0.f, 0.f};
  for (int k0 = 0; k0 < Hn; k0 += 32) {
    __builtin_prefetch(Whh + (size_t)brow0 * Hn + k0 + 256, 0, 3);
    __builtin_prefetch(Whh + (size_t)brow1 * Hn + k0 + 256, 0, 3);
    FragU a, b0, b1;
    load_frag(a, sh, Hn, lane & 15, k0, lane);
    load_frag(b0, Whh, Hn, brow0, k0, lane);
    load_frag(b1, Whh, Hn, brow1, k0, lane);
    acc0 = __builtin_amdgcn_wmma_f32_16x16x32_bf16(false, a.v, false, b0.v,
                                                   (short)0, acc0, false, false);
    acc1 = __builtin_amdgcn_wmma_f32_16x16x32_bf16(false, a.v, false, b1.v,
                                                   (short)0, acc1, false, false);
  }
  {
    const int nn = lane & 15;
    const int mb = ((lane >> 4) & 1) * 8;
#pragma unroll
    for (int v = 0; v < 8; ++v) {
      int bi = mb + v;
      sg[g][bi][jj * 16 + nn] =
          acc0[v] + G[((size_t)bi * Tn + t) * G4 + n0t + nn];
      sg[g][bi][(jj + 1) * 16 + nn] =
          acc1[v] + G[((size_t)bi * Tn + t) * G4 + n0t + 16 + nn];
    }
  }
  __syncthreads();

  for (int idx = threadIdx.x; idx < Bn * 64; idx += blockDim.x) {
    int bi = idx >> 6, col = idx & 63;
    int hc = c0 + col;
    float ig = sigmoidf_(sg[0][bi][col]);
    float fg = sigmoidf_(sg[1][bi][col]);
    float gg = tanhf(sg[2][bi][col]);
    float og = sigmoidf_(sg[3][bi][col]);
    float cold = cst[bi * Hn + hc];
    float cnew = fg * cold + ig * gg;
    float hnew = og * tanhf(cnew);
    cst[bi * Hn + hc] = cnew;
    hwrite[bi * Hn + hc] = f2bf(hnew);
    hout[((size_t)bi * Tn + t) * Hn + hc] = hnew;
  }
}

// ---------------------------------------------------------------------------
// Combine dirs into next-layer input (bf16) + final feats (fp32), applying the
// mask and un-reversing the backward outputs.
__global__ void k_gather(const float* __restrict__ hF, const float* __restrict__ hB,
                         const float* __restrict__ mask, const int* __restrict__ len,
                         unsigned short* __restrict__ xnext, float* __restrict__ feats,
                         int last) {
  int idx = blockIdx.x * blockDim.x + threadIdx.x;
  if (idx >= Bn * Tn * Hn) return;
  int d = idx % Hn; int r = idx / Hn; int t = r % Tn; int b = r / Tn;
  float m = mask[b * Tn + t];
  int lb = len[b]; int tr = (t < lb) ? (lb - 1 - t) : t;
  float vf = hF[((size_t)b * Tn + t) * Hn + d] * m;
  float vb = hB[((size_t)b * Tn + tr) * Hn + d] * m;
  size_t o = ((size_t)b * Tn + t) * (2 * Hn);
  xnext[o + d] = f2bf(vf);
  xnext[o + Hn + d] = f2bf(vb);
  if (last) { feats[o + d] = vf; feats[o + Hn + d] = vb; }
}

__global__ void k_logits(const float* __restrict__ feats, const float* __restrict__ lw,
                         const float* __restrict__ lb, float* __restrict__ logits) {
  int row = blockIdx.x;
  int k = threadIdx.x;
  if (k >= Kc) return;
  const float* f = feats + (size_t)row * (2 * Hn);
  const float* w = lw + (size_t)k * (2 * Hn);
  float s = lb[k];
  for (int i = 0; i < 2 * Hn; ++i) s += f[i] * w[i];
  logits[(size_t)row * Kc + k] = s;
}

// ---------------------------------------------------------------------------
// CRF: numerator + forward logsumexp + Viterbi + backtrace + loss. One block.
__global__ void k_crf(const float* __restrict__ logits, const float* __restrict__ mask,
                      const int* __restrict__ len, const int* __restrict__ labels,
                      const float* __restrict__ trans, const float* __restrict__ startv,
                      const float* __restrict__ endv, int* __restrict__ bp,
                      float* __restrict__ out) {
  __shared__ float tr_s[Kc * Kc], st_s[Kc], en_s[Kc];
  __shared__ float alpha[Bn][Kc], va[Bn][Kc];
  __shared__ float num_s[Bn], den_s[Bn];
  int tid = threadIdx.x;
  if (tid < Kc * Kc) tr_s[tid] = trans[tid];
  if (tid < Kc) { st_s[tid] = startv[tid]; en_s[tid] = endv[tid]; }
  __syncthreads();

  if (tid < Bn) {
    int b = tid, lb = len[b];
    float s = 0.f; int prev = 0;
    for (int t = 0; t < Tn; ++t) {
      int lab = labels[b * Tn + t]; int tag = (lab == -100) ? 0 : lab;
      if (mask[b * Tn + t] > 0.5f) {
        s += logits[(b * Tn + t) * Kc + tag];
        if (t > 0) s += tr_s[prev * Kc + tag];
      }
      prev = tag;
    }
    int t0 = labels[b * Tn]; if (t0 == -100) t0 = 0;
    int tl = labels[b * Tn + lb - 1]; if (tl == -100) tl = 0;
    num_s[b] = st_s[t0] + s + en_s[tl];
  }

  bool act = tid < Bn * Kc;
  int b = tid / Kc, j = tid % Kc;
  if (act) {
    float a0 = st_s[j] + logits[(b * Tn) * Kc + j];
    alpha[b][j] = a0; va[b][j] = a0;
  }
  __syncthreads();

  for (int t = 1; t < Tn; ++t) {
    float na = 0.f, nv = 0.f; int bi = j; bool upd = false;
    if (act) {
      upd = mask[b * Tn + t] > 0.5f;
      float e = logits[(b * Tn + t) * Kc + j];
      float mx = -1e30f;
      for (int i = 0; i < Kc; ++i) mx = fmaxf(mx, alpha[b][i] + tr_s[i * Kc + j]);
      float ss = 0.f;
      for (int i = 0; i < Kc; ++i) ss += __expf(alpha[b][i] + tr_s[i * Kc + j] - mx);
      na = mx + __logf(ss) + e;
      float best = -1e30f; bi = 0;
      for (int i = 0; i < Kc; ++i) {
        float v = va[b][i] + tr_s[i * Kc + j];
        if (v > best) { best = v; bi = i; }
      }
      nv = best + e;
    }
    __syncthreads();
    if (act) {
      if (upd) { alpha[b][j] = na; va[b][j] = nv; bp[(t - 1) * (Bn * Kc) + tid] = bi; }
      else bp[(t - 1) * (Bn * Kc) + tid] = j;
    }
    __syncthreads();
  }

  if (tid < Bn) {
    int bb = tid;
    float mx = -1e30f;
    for (int k = 0; k < Kc; ++k) mx = fmaxf(mx, alpha[bb][k] + en_s[k]);
    float ss = 0.f;
    for (int k = 0; k < Kc; ++k) ss += __expf(alpha[bb][k] + en_s[k] - mx);
    den_s[bb] = mx + __logf(ss);
    float best = -1e30f; int cur = 0;
    for (int k = 0; k < Kc; ++k) {
      float v = va[bb][k] + en_s[k];
      if (v > best) { best = v; cur = k; }
    }
    out[1 + bb * Tn + (Tn - 1)] = (mask[bb * Tn + Tn - 1] > 0.5f) ? (float)cur : -100.f;
    for (int s = Tn - 2; s >= 0; --s) {
      cur = bp[s * (Bn * Kc) + bb * Kc + cur];
      out[1 + bb * Tn + s] = (mask[bb * Tn + s] > 0.5f) ? (float)cur : -100.f;
    }
  }
  __syncthreads();
  if (tid == 0) {
    float L = 0.f;
    for (int k = 0; k < Bn; ++k) L += num_s[k] - den_s[k];
    out[0] = -L;
  }
}

// ---------------------------------------------------------------------------
extern "C" void kernel_launch(void* const* d_in, const int* in_sizes, int n_in,
                              void* d_out, int out_size, void* d_ws, size_t ws_size,
                              hipStream_t stream) {
  (void)in_sizes; (void)n_in; (void)out_size; (void)ws_size;
  const float* x = (const float*)d_in[0];
  const float* mask = (const float*)d_in[1];
  const int* labels = (const int*)d_in[2];
  const float *wih[12], *whh[12], *bih[12], *bhh[12];
  for (int p = 0; p < 12; ++p) {
    int b0 = 3 + 4 * p;
    wih[p] = (const float*)d_in[b0];
    whh[p] = (const float*)d_in[b0 + 1];
    bih[p] = (const float*)d_in[b0 + 2];
    bhh[p] = (const float*)d_in[b0 + 3];
  }
  const float* lw = (const float*)d_in[51];
  const float* lbv = (const float*)d_in[52];
  const float* trans = (const float*)d_in[53];
  const float* startv = (const float*)d_in[54];
  const float* endv = (const float*)d_in[55];

  char* ws = (char*)d_ws;
  size_t cur = 0;
  auto alloc = [&](size_t bytes) -> char* {
    cur = (cur + 255) & ~(size_t)255;
    char* p = ws + cur; cur += bytes; return p;
  };
  auto cdiv = [](int n, int b) { return (n + b - 1) / b; };

  int* lengths = (int*)alloc(Bn * sizeof(int));
  float* biasS[12]; unsigned short* wihB[12]; unsigned short* whhB[12];
  for (int p = 0; p < 12; ++p) {
    int ind = (p < 2) ? Dn : 2 * Hn;
    biasS[p] = (float*)alloc((size_t)G4 * 4);
    wihB[p] = (unsigned short*)alloc((size_t)G4 * ind * 2);
    whhB[p] = (unsigned short*)alloc((size_t)G4 * Hn * 2);
  }
  unsigned short* x0f = (unsigned short*)alloc((size_t)Bn * Tn * Dn * 2);
  unsigned short* x0b = (unsigned short*)alloc((size_t)Bn * Tn * Dn * 2);
  unsigned short* x1f = (unsigned short*)alloc((size_t)Bn * Tn * 2 * Hn * 2);
  unsigned short* x1b = (unsigned short*)alloc((size_t)Bn * Tn * 2 * Hn * 2);
  float* Gf = (float*)alloc((size_t)Bn * Tn * G4 * 4);
  float* Gb = (float*)alloc((size_t)Bn * Tn * G4 * 4);
  unsigned short* hbuf = (unsigned short*)alloc((size_t)2 * 2 * Bn * Hn * 2);
  float* cst = (float*)alloc((size_t)2 * Bn * Hn * 4);
  float* houtF = (float*)alloc((size_t)Bn * Tn * Hn * 4);
  float* houtB = (float*)alloc((size_t)Bn * Tn * Hn * 4);
  float* feats = (float*)alloc((size_t)Bn * Tn * 2 * Hn * 4);
  float* logits = (float*)alloc((size_t)Bn * Tn * Kc * 4);
  int* bp = (int*)alloc((size_t)(Tn - 1) * Bn * Kc * 4);

  k_lengths<<<1, 32, 0, stream>>>(mask, lengths);
  for (int p = 0; p < 12; ++p) {
    int ind = (p < 2) ? Dn : 2 * Hn;
    int n1 = G4 * ind;
    k_f2bf<<<cdiv(n1, 256), 256, 0, stream>>>(wih[p], wihB[p], n1);
    k_f2bf<<<cdiv(G4 * Hn, 256), 256, 0, stream>>>(whh[p], whhB[p], G4 * Hn);
    k_bias<<<cdiv(G4, 256), 256, 0, stream>>>(bih[p], bhh[p], biasS[p]);
  }
  k_f2bf<<<cdiv(Bn * Tn * Dn, 256), 256, 0, stream>>>(x, x0f, Bn * Tn * Dn);
  k_reverse<<<cdiv(Bn * Tn * Dn, 256), 256, 0, stream>>>(x0f, x0b, lengths, Dn);

  const unsigned short* xf = x0f;
  const unsigned short* xb = x0b;
  for (int layer = 0; layer < 3; ++layer) {
    int ind = layer ? 2 * Hn : Dn;
    int pf = 2 * layer, pb = pf + 1;
    size_t smem = (size_t)16 * ind * 2;
    k_input_gemm<<<dim3(Bn * Tn / 16, G4 / 256), 256, smem, stream>>>(
        xf, wihB[pf], biasS[pf], Gf, ind);
    k_input_gemm<<<dim3(Bn * Tn / 16, G4 / 256), 256, smem, stream>>>(
        xb, wihB[pb], biasS[pb], Gb, ind);
    k_zero_state<<<cdiv(2 * 2 * Bn * Hn, 256), 256, 0, stream>>>(hbuf, cst);
    for (int t = 0; t < Tn; ++t) {
      k_lstm_step<<<dim3(Hn / 64, 2), 256, 0, stream>>>(
          whhB[pf], whhB[pb], Gf, Gb, hbuf, cst, houtF, houtB, t);
    }
    k_gather<<<cdiv(Bn * Tn * Hn, 256), 256, 0, stream>>>(
        houtF, houtB, mask, lengths, x1f, feats, (layer == 2) ? 1 : 0);
    k_reverse<<<cdiv(Bn * Tn * 2 * Hn, 256), 256, 0, stream>>>(x1f, x1b, lengths, 2 * Hn);
    xf = x1f; xb = x1b;
  }
  k_logits<<<Bn * Tn, 32, 0, stream>>>(feats, lw, lbv, logits);
  k_crf<<<1, 256, 0, stream>>>(logits, mask, lengths, labels, trans, startv, endv,
                               bp, (float*)d_out);
}
